// MultiheadLatentAttention_30588757082263
// MI455X (gfx1250) — matrix operand order repaired and checked by reading, not compiled, not run on hardware
//
#include <hip/hip_runtime.h>
#include <cstddef>

typedef __bf16 bf16;
typedef __attribute__((ext_vector_type(16))) __bf16 v16bf;
typedef __attribute__((ext_vector_type(8)))  __bf16 v8bf;
typedef __attribute__((ext_vector_type(8)))  float  v8f;

union V16U { v16bf v; v8bf h[2]; };

#define WMMA_BF16(a, b, c) \
  __builtin_amdgcn_wmma_f32_16x16x32_bf16(false, (a), false, (b), (short)0, (c), false, false)

// WGP-scope prefetch (scope 0 = default): pull into all cache levels incl. WGP$.
// (__builtin_prefetch only reaches SE/SYS scope, which stops at GL2.)
#define PREFETCH_WGP(ptr, imm_bytes) \
  asm volatile("global_prefetch_b8 %0, off offset:" #imm_bytes :: "v"(ptr))

// Raw cross-lane gather: dst[lane] = src[idx_bytes/4]  (one ds_bpermute_b32,
// no width-clamp VALU that the __shfl_xor wrapper emits on wave32).
__device__ __forceinline__ float bperm_f32(int idx_bytes, float v) {
  return __int_as_float(__builtin_amdgcn_ds_bpermute(idx_bytes, __float_as_int(v)));
}

// ---------------------------------------------------------------------------
// f32 -> bf16 elementwise convert
// ---------------------------------------------------------------------------
__global__ __launch_bounds__(256) void f32_to_bf16_kernel(const float* __restrict__ in,
                                                          bf16* __restrict__ out, long n) {
  long i = (long)blockIdx.x * blockDim.x + threadIdx.x;
  if (i < n) out[i] = (bf16)in[i];
}

// f32 [R][C] -> bf16 transposed [C][R]
__global__ __launch_bounds__(256) void f32_to_bf16_T_kernel(const float* __restrict__ in,
                                                            bf16* __restrict__ out,
                                                            int R, int C) {
  long i = (long)blockIdx.x * blockDim.x + threadIdx.x;
  long n = (long)R * C;
  if (i < n) {
    int r = (int)(i / C);
    int c = (int)(i % C);
    out[(size_t)c * R + r] = (bf16)in[i];
  }
}

// ---------------------------------------------------------------------------
// WMMA GEMM: C[M,N] = A[M,K] * B[K,N], A row-major bf16, BT = B^T [N,K] bf16.
// Block: 256 threads = 8 waves (4x2), block tile 128x128,
// wave tile 32x64 (2 A-frags x 4 B-frags -> 8 WMMAs / K-step), K step 32.
// TRANS_OUT=true writes C transposed as [N,M].
// ---------------------------------------------------------------------------
template <typename OutT, bool TRANS_OUT>
__global__ __launch_bounds__(256) void wmma_gemm_bf16(const bf16* __restrict__ A,
                                                      const bf16* __restrict__ BT,
                                                      OutT* __restrict__ C,
                                                      int M, int N, int K) {
  const int lane = threadIdx.x & 31;
  const int wave = threadIdx.x >> 5;
  const int col  = lane & 15;   // B/C fragment column, A fragment row
  const int half = lane >> 4;   // lane group 0: lanes 0-15, 1: lanes 16-31
  const int m0 = blockIdx.y * 128 + (wave >> 1) * 32;
  const int n0 = blockIdx.x * 128 + (wave & 1) * 64;

  // Base pointers (bumped by 32 each K-step to avoid 64-bit addr recompute)
  const bf16* ap[2];
#pragma unroll
  for (int i = 0; i < 2; ++i)
    ap[i] = A + (size_t)(m0 + i * 16 + col) * K + half * 16;
  const bf16* bp[4];
#pragma unroll
  for (int j = 0; j < 4; ++j)
    bp[j] = BT + (size_t)(n0 + j * 16 + col) * K + half * 16;

  v8f acc[2][4] = {};

#pragma unroll 2
  for (int k0 = 0; k0 < K; k0 += 32) {
    V16U ua[2], ub[4];
#pragma unroll
    for (int i = 0; i < 2; ++i) {
      // A fragment 16x32: lane<16 holds K 0-15, lane>=16 holds K 16-31
      ua[i].h[0] = *(const v8bf*)(ap[i]);
      ua[i].h[1] = *(const v8bf*)(ap[i] + 8);
      // WGP-scope prefetch of the K-tile 4 steps ahead (256 B = 128 elems)
      PREFETCH_WGP(ap[i], 256);
      ap[i] += 32;
    }
#pragma unroll
    for (int j = 0; j < 4; ++j) {
      // B fragment 32x16 via BT: lanes 0-15 hold K 0-15, lanes 16-31 hold K 16-31
      ub[j].v = *(const v16bf*)(bp[j]);
      PREFETCH_WGP(bp[j], 256);
      bp[j] += 32;
    }
#pragma unroll
    for (int i = 0; i < 2; ++i)
#pragma unroll
      for (int j = 0; j < 4; ++j)
        acc[i][j] = WMMA_BF16(ua[i].v, ub[j].v, acc[i][j]);
  }

#pragma unroll
  for (int i = 0; i < 2; ++i)
#pragma unroll
    for (int j = 0; j < 4; ++j)
#pragma unroll
      for (int e = 0; e < 8; ++e) {
        const int m = m0 + i * 16 + half * 8 + e;
        const int n = n0 + j * 16 + col;
        const float v = acc[i][j][e];
        if (TRANS_OUT) C[(size_t)n * M + m] = (OutT)v;
        else           C[(size_t)m * N + n] = (OutT)v;
      }
}

// ---------------------------------------------------------------------------
// Flash attention: Q,K in [B*S, H*D] bf16 row-major; Vt in [H*D, B*S] bf16.
// Grid (S/128, H, B); block 256 = 8 waves; each wave owns 16 query rows.
// ---------------------------------------------------------------------------
__global__ __launch_bounds__(256) void mla_attention_kernel(const bf16* __restrict__ Q,
                                                            const bf16* __restrict__ Kmat,
                                                            const bf16* __restrict__ Vt,
                                                            bf16* __restrict__ O) {
  constexpr int S = 2048, E = 2048, D = 128, BS = 4096;
  constexpr float scale = 0.08838834764831845f;  // 1/sqrt(128)

  __shared__ bf16 plds[8][16 * 32];  // per-wave private P staging (16 q-rows x 32 keys)

  const int lane = threadIdx.x & 31;
  const int wave = threadIdx.x >> 5;
  const int col  = lane & 15;
  const int half = lane >> 4;
  const int b  = blockIdx.z;
  const int h  = blockIdx.y;
  const int q0 = blockIdx.x * 128 + wave * 16;

  // Butterfly bpermute byte-indices (loop-invariant).  Masks 1,2,4,8 stay
  // inside each 16-lane row group, so wave-wrap bpermute is exact.
  int xidx[4];
#pragma unroll
  for (int t = 0; t < 4; ++t) xidx[t] = (lane ^ (1 << t)) << 2;

  // Q fragments for K-dim chunks d = 0..127 in steps of 32 (kept in registers)
  V16U qf[4];
  {
    const bf16* qrow = Q + ((size_t)(b * S + q0 + col)) * E + h * D;
#pragma unroll
    for (int dk = 0; dk < 4; ++dk) {
      qf[dk].h[0] = *(const v8bf*)(qrow + dk * 32 + half * 16);
      qf[dk].h[1] = *(const v8bf*)(qrow + dk * 32 + half * 16 + 8);
    }
  }

  // Streaming base pointers, bumped by one 32-key block per iteration
  const bf16* kp0 = Kmat + ((size_t)(b * S + col)) * E      + h * D + half * 16;  // keys j..j+15
  const bf16* kp1 = Kmat + ((size_t)(b * S + 16 + col)) * E + h * D + half * 16;  // keys j+16..j+31
  const bf16* vp[8];
#pragma unroll
  for (int dt = 0; dt < 8; ++dt)
    vp[dt] = Vt + ((size_t)(h * D + dt * 16 + col)) * BS + (size_t)b * S + half * 16;

  float mrun[8], lrun[8];
  v8f acc[8] = {};
#pragma unroll
  for (int e = 0; e < 8; ++e) { mrun[e] = -1e30f; lrun[e] = 0.0f; }

  for (int j = 0; j < S; j += 32) {
    // ---- scores: two 16x16 tiles (keys j..j+15, j+16..j+31) ----
    v8f s0 = {}, s1 = {};
#pragma unroll
    for (int dk = 0; dk < 4; ++dk) {
      const v16bf kf0 = *(const v16bf*)(kp0 + dk * 32);
      const v16bf kf1 = *(const v16bf*)(kp1 + dk * 32);
      s0 = WMMA_BF16(qf[dk].v, kf0, s0);
      s1 = WMMA_BF16(qf[dk].v, kf1, s1);
    }
    // WGP-scope prefetch of the next key block (32 rows * E * 2B = 131072 B ahead)
    PREFETCH_WGP(kp0, 131072);
    PREFETCH_WGP(kp1, 131072);
    kp0 += (size_t)32 * E;
    kp1 += (size_t)32 * E;

    // ---- online softmax over the 32-key block ----
    float alpha[8];
#pragma unroll
    for (int e = 0; e < 8; ++e) {
      const float a0 = s0[e] * scale;
      const float a1 = s1[e] * scale;
      float rmax = fmaxf(a0, a1);
#pragma unroll
      for (int t = 0; t < 4; ++t)
        rmax = fmaxf(rmax, bperm_f32(xidx[t], rmax));
      const float mnew = fmaxf(mrun[e], rmax);
      const float al = __expf(mrun[e] - mnew);
      const float p0 = __expf(a0 - mnew);
      const float p1 = __expf(a1 - mnew);
      float rsum = p0 + p1;
#pragma unroll
      for (int t = 0; t < 4; ++t)
        rsum += bperm_f32(xidx[t], rsum);
      lrun[e] = lrun[e] * al + rsum;
      mrun[e] = mnew;
      alpha[e] = al;
      // write P (C-layout -> row-major LDS tile [row][k])
      plds[wave][(half * 8 + e) * 32 + col]      = (bf16)p0;
      plds[wave][(half * 8 + e) * 32 + 16 + col] = (bf16)p1;
    }

    // rescale accumulator rows
#pragma unroll
    for (int dt = 0; dt < 8; ++dt)
#pragma unroll
      for (int e = 0; e < 8; ++e)
        acc[dt][e] *= alpha[e];

    // wave-private LDS: DS ops are in-order per wave; split-counter wait
    asm volatile("s_wait_dscnt 0" ::: "memory");

    // re-read P as a 16x32 A-fragment
    V16U pa;
    pa.h[0] = *(const v8bf*)&plds[wave][col * 32 + half * 16];
    pa.h[1] = *(const v8bf*)&plds[wave][col * 32 + 8 + half * 16];

    // ---- P x V: 8 output d-tiles of 16, each streaming its own row ----
#pragma unroll
    for (int dt = 0; dt < 8; ++dt) {
      const v16bf vb = *(const v16bf*)(vp[dt]);
      PREFETCH_WGP(vp[dt], 64);  // next 32 keys of this V row
      acc[dt] = WMMA_BF16(pa.v, vb, acc[dt]);
      vp[dt] += 32;
    }
  }

  // epilogue: divide by softmax denominator, write bf16 [B*S, H*D]
#pragma unroll
  for (int e = 0; e < 8; ++e) {
    const float inv = 1.0f / lrun[e];
    const size_t rowoff = ((size_t)(b * S + q0 + half * 8 + e)) * E + h * D;
#pragma unroll
    for (int dt = 0; dt < 8; ++dt)
      O[rowoff + dt * 16 + col] = (bf16)(acc[dt][e] * inv);
  }
}

// ---------------------------------------------------------------------------
// Host launch
// ---------------------------------------------------------------------------
extern "C" void kernel_launch(void* const* d_in, const int* in_sizes, int n_in,
                              void* d_out, int out_size, void* d_ws, size_t ws_size,
                              hipStream_t stream) {
  (void)in_sizes; (void)n_in; (void)out_size; (void)ws_size;
  constexpr int Bz = 2, S = 2048, E = 2048, L = 512, H = 16;
  constexpr int M = Bz * S;  // 4096

  const float* x   = (const float*)d_in[0];
  const float* W_q = (const float*)d_in[1];
  const float* W_c = (const float*)d_in[2];
  const float* W_k = (const float*)d_in[3];
  const float* W_v = (const float*)d_in[4];
  const float* W_o = (const float*)d_in[5];

  char* ws = (char*)d_ws;
  size_t off = 0;
  auto alloc = [&](size_t elems) {
    char* p = ws + off;
    off += (elems * sizeof(bf16) + 255) & ~(size_t)255;
    return (bf16*)p;
  };
  bf16* x_bf  = alloc((size_t)M * E);
  bf16* WqT   = alloc((size_t)E * E);   // [E][E]  (B^T of W_q)
  bf16* WcT   = alloc((size_t)L * E);   // [L][E]
  bf16* WkT   = alloc((size_t)E * L);   // [E][L]
  bf16* WvT   = alloc((size_t)E * L);   // [E][L]
  bf16* WoT   = alloc((size_t)E * E);   // [E][E]
  bf16* Qb    = alloc((size_t)M * E);
  bf16* Cb    = alloc((size_t)M * L);
  bf16* Kb    = alloc((size_t)M * E);
  bf16* Vt    = alloc((size_t)E * M);   // [H*D][B*S]
  bf16* attnb = alloc((size_t)M * E);

  auto cdiv = [](long a, long b) { return (unsigned)((a + b - 1) / b); };

  // Convert inputs to bf16 (weights transposed for contiguous B fragments)
  f32_to_bf16_kernel<<<cdiv((long)M * E, 256), 256, 0, stream>>>(x, x_bf, (long)M * E);
  f32_to_bf16_T_kernel<<<cdiv((long)E * E, 256), 256, 0, stream>>>(W_q, WqT, E, E);
  f32_to_bf16_T_kernel<<<cdiv((long)E * L, 256), 256, 0, stream>>>(W_c, WcT, E, L);
  f32_to_bf16_T_kernel<<<cdiv((long)L * E, 256), 256, 0, stream>>>(W_k, WkT, L, E);
  f32_to_bf16_T_kernel<<<cdiv((long)L * E, 256), 256, 0, stream>>>(W_v, WvT, L, E);
  f32_to_bf16_T_kernel<<<cdiv((long)E * E, 256), 256, 0, stream>>>(W_o, WoT, E, E);

  // Projections (block tile 128x128)
  wmma_gemm_bf16<bf16, false><<<dim3(E / 128, M / 128), 256, 0, stream>>>(x_bf, WqT, Qb, M, E, E);
  wmma_gemm_bf16<bf16, false><<<dim3(L / 128, M / 128), 256, 0, stream>>>(x_bf, WcT, Cb, M, L, E);
  wmma_gemm_bf16<bf16, false><<<dim3(E / 128, M / 128), 256, 0, stream>>>(Cb, WkT, Kb, M, E, L);
  wmma_gemm_bf16<bf16, true ><<<dim3(E / 128, M / 128), 256, 0, stream>>>(Cb, WvT, Vt, M, E, L);

  // Attention (flash, per head)
  mla_attention_kernel<<<dim3(S / 128, H, Bz), 256, 0, stream>>>(Qb, Kb, Vt, attnb);

  // Output projection -> f32
  wmma_gemm_bf16<float, false><<<dim3(E / 128, M / 128), 256, 0, stream>>>(
      attnb, WoT, (float*)d_out, M, E, E);
}
